// Model_1838246003034
// MI455X (gfx1250) — matrix operand order repaired and checked by reading, not compiled, not run on hardware
//
#include <hip/hip_runtime.h>
#include <math.h>

// ---------------- problem constants (match reference) ----------------
#define Hc    128
#define EDc   64
#define Lc    2
#define NUc   100000
#define NBc   50000
#define NGc   500
#define EUBc  500000
#define EBGc  100000
#define NLc   100000
#define SCALEc 0.08838834764831845f   // 1/sqrt(128)

// host-side constant: fenc(-INFINITY) == ~0xFF800000 == 0x007FFFFF
#define FENC_NEG_INF 0x007FFFFFu

// ---------------- WMMA types ----------------
typedef __bf16 v16bf __attribute__((ext_vector_type(16)));
typedef float  v8f   __attribute__((ext_vector_type(8)));
union FragAB { v16bf bf; unsigned u[8]; };

// f32 -> bf16 round-to-nearest-even (bit trick, no __bf16 arithmetic needed)
__device__ __forceinline__ unsigned short f2bf(float f) {
  unsigned u = __float_as_uint(f);
  u += 0x7FFFu + ((u >> 16) & 1u);
  return (unsigned short)(u >> 16);
}
// order-preserving float<->uint mapping for atomicMax-based segment max
__device__ __forceinline__ unsigned fenc(float f) {
  unsigned u = __float_as_uint(f);
  return (u & 0x80000000u) ? ~u : (u | 0x80000000u);
}
__device__ __forceinline__ float fdec(unsigned u) {
  unsigned v = (u & 0x80000000u) ? (u & 0x7FFFFFFFu) : ~u;
  return __uint_as_float(v);
}

// =====================================================================
// GEMM: C[M,128] (=|+=) A[M,K] @ W[K,128] + bias.   K in {64,128}.
// bf16 inputs staged in LDS, f32 WMMA accumulate. 256 thr = 8 waves.
// Block tile: 64(M) x 128(N); wave w owns columns [16w,16w+16).
// =====================================================================
__global__ __launch_bounds__(256) void gemm_bf16_wmma(
    const float* __restrict__ A, const float* __restrict__ W,
    const float* __restrict__ bias, float* __restrict__ C,
    int M, int K, int accum)
{
  __shared__ unsigned short sA [64  * 136];  // A tile, row-major, padded
  __shared__ unsigned short sWt[128 * 136];  // W transposed: sWt[n][k]
  const int tid = threadIdx.x;
  const int m0  = blockIdx.x * 64;

  // stage entire weight (K*128) transposed into LDS, converted to bf16
  for (int i = tid; i < K * 128; i += 256) {
    int k = i >> 7, n = i & 127;
    sWt[n * 136 + k] = f2bf(W[i]);
  }
  // stage A tile (64 x K)
  for (int i = tid; i < 64 * K; i += 256) {
    int r = i / K, c = i - r * K;
    int gr = m0 + r;
    float v = (gr < M) ? A[(size_t)gr * K + c] : 0.f;
    sA[r * 136 + c] = f2bf(v);
  }
  __syncthreads();

  const int wave = tid >> 5, lane = tid & 31;
  const int hl = lane & 15, hi = lane >> 4;
  const int ncol = wave * 16 + hl;            // this lane's output column
  const unsigned short* bcol = &sWt[ncol * 136];

  for (int mt = 0; mt < 4; ++mt) {
    v8f acc = {0.f, 0.f, 0.f, 0.f, 0.f, 0.f, 0.f, 0.f};
    const unsigned short* arow = &sA[(mt * 16 + hl) * 136];
    for (int ks = 0; ks < K; ks += 32) {
      FragAB a, b;
#pragma unroll
      for (int p = 0; p < 8; ++p) {
        // A 16x32 bf16 layout: lane<16 holds K {0..7,16..23}, lane>=16 {+8}
        int ka = ks + ((p < 4) ? (2 * p) : (2 * p + 8)) + hi * 8;
        a.u[p] = *(const unsigned*)(arow + ka);
        // B 32x16: same K layout as A: lane<16 holds K {0..7,16..23}, lane>=16 {+8}
        int kb = ks + ((p < 4) ? (2 * p) : (2 * p + 8)) + hi * 8;
        b.u[p] = *(const unsigned*)(bcol + kb);
      }
      acc = __builtin_amdgcn_wmma_f32_16x16x32_bf16(
          false, a.bf, false, b.bf, (short)0, acc, false, false);
    }
    float bv = bias ? bias[ncol] : 0.f;
#pragma unroll
    for (int r = 0; r < 8; ++r) {            // C/D: row = r + 8*hi, col = hl
      int row = m0 + mt * 16 + r + hi * 8;
      if (row < M) {
        float val = acc[r] + bv;
        size_t o = (size_t)row * 128 + ncol;
        if (accum) C[o] += val; else C[o] = val;
      }
    }
  }
}

// =====================================================================
// elementwise / gather helpers
// =====================================================================
__global__ void fill_f32(float* p, float v, int n) {
  int i = blockIdx.x * blockDim.x + threadIdx.x;
  if (i < n) p[i] = v;
}
__global__ void fill_u32(unsigned* p, unsigned v, int n) {
  int i = blockIdx.x * blockDim.x + threadIdx.x;
  if (i < n) p[i] = v;
}
__global__ void gather_rows(const float* __restrict__ emb, const int* __restrict__ idx,
                            float* __restrict__ out, int n) {
  int i = blockIdx.x * blockDim.x + threadIdx.x;   // over n*128
  if (i < n * Hc) {
    int r = i >> 7, c = i & 127;
    out[i] = emb[(size_t)idx[r] * Hc + c];
  }
}
__global__ void relu_k(float* p, int n) {
  int i = blockIdx.x * blockDim.x + threadIdx.x;
  if (i < n) p[i] = fmaxf(p[i], 0.f);
}

// =====================================================================
// edge kernels: one wave (32 lanes) per edge; lane handles 4 features
// =====================================================================
__global__ __launch_bounds__(256) void edge_alpha(
    const float* __restrict__ q, const float* __restrict__ k,
    const float* __restrict__ e,
    const int* __restrict__ src, const int* __restrict__ dst, int E,
    float* __restrict__ alpha, unsigned* __restrict__ m_u)
{
  int eid = blockIdx.x * 8 + (threadIdx.x >> 5);
  if (eid >= E) return;
  int lane = threadIdx.x & 31;
  int d = dst[eid], s = src[eid];
  const float4 qv = *(const float4*)&q[(size_t)d * Hc + lane * 4];
  const float4 kv = *(const float4*)&k[(size_t)s * Hc + lane * 4];
  const float4 ev = *(const float4*)&e[(size_t)eid * Hc + lane * 4];
  float p = qv.x * (kv.x + ev.x) + qv.y * (kv.y + ev.y) +
            qv.z * (kv.z + ev.z) + qv.w * (kv.w + ev.w);
  for (int off = 16; off; off >>= 1) p += __shfl_xor(p, off, 32);
  if (lane == 0) {
    float a = p * SCALEc;
    alpha[eid] = a;
    atomicMax(&m_u[d], fenc(a));
  }
}

__global__ void edge_exp(const float* __restrict__ alpha,
                         const int* __restrict__ dst, int E,
                         const unsigned* __restrict__ m_u,
                         float* __restrict__ w, float* __restrict__ z)
{
  int eid = blockIdx.x * blockDim.x + threadIdx.x;
  if (eid >= E) return;
  int d = dst[eid];
  float ww = __expf(alpha[eid] - fdec(m_u[d]));
  w[eid] = ww;
  atomicAdd(&z[d], ww);
}

__global__ __launch_bounds__(256) void edge_scatter(
    const float* __restrict__ v, const float* __restrict__ e,
    const int* __restrict__ src, const int* __restrict__ dst, int E,
    const float* __restrict__ w, const float* __restrict__ z,
    float* __restrict__ out)
{
  int eid = blockIdx.x * 8 + (threadIdx.x >> 5);
  if (eid >= E) return;
  int lane = threadIdx.x & 31;
  int d = dst[eid], s = src[eid];
  float coeff = w[eid] / z[d];
  const float4 vv = *(const float4*)&v[(size_t)s * Hc + lane * 4];
  const float4 ev = *(const float4*)&e[(size_t)eid * Hc + lane * 4];
  float* o = &out[(size_t)d * Hc + lane * 4];
  atomicAdd(o + 0, (vv.x + ev.x) * coeff);
  atomicAdd(o + 1, (vv.y + ev.y) * coeff);
  atomicAdd(o + 2, (vv.z + ev.z) * coeff);
  atomicAdd(o + 3, (vv.w + ev.w) * coeff);
}

__global__ __launch_bounds__(256) void label_dot(
    const float* __restrict__ xu, const float* __restrict__ xb,
    const int* __restrict__ ls, const int* __restrict__ ld,
    float* __restrict__ out, int n)
{
  int i = blockIdx.x * 8 + (threadIdx.x >> 5);
  if (i >= n) return;
  int lane = threadIdx.x & 31;
  const float4 a = *(const float4*)&xu[(size_t)ls[i] * Hc + lane * 4];
  const float4 b = *(const float4*)&xb[(size_t)ld[i] * Hc + lane * 4];
  float p = a.x * b.x + a.y * b.y + a.z * b.z + a.w * b.w;
  for (int off = 16; off; off >>= 1) p += __shfl_xor(p, off, 32);
  if (lane == 0) out[i] = p;
}

// =====================================================================
// host orchestration
// =====================================================================
struct Scratch {
  float *qb, *kb, *vb, *eb, *al, *wb, *zb;
  unsigned* mu;
};

static void gemm(const float* A, const float* W, const float* bias, float* C,
                 int M, int K, int accum, hipStream_t s) {
  gemm_bf16_wmma<<<(M + 63) / 64, 256, 0, s>>>(A, W, bias, C, M, K, accum);
}

static void run_tconv(const float* xsrc, const float* xdst, int n_src, int n_dst,
                      const int* src, const int* dst, int E, const float* ea,
                      const float* Wq, const float* bq,
                      const float* Wk, const float* bk,
                      const float* Wv, const float* bv,
                      const float* We,
                      const float* Ws, const float* bs,
                      float* out, int accum_skip,
                      const Scratch& S, hipStream_t s)
{
  gemm(xdst, Wq, bq, S.qb, n_dst, Hc, 0, s);                    // q
  gemm(xsrc, Wk, bk, S.kb, n_src, Hc, 0, s);                    // k
  gemm(xsrc, Wv, bv, S.vb, n_src, Hc, 0, s);                    // v
  gemm(ea,   We, nullptr, S.eb, E, EDc, 0, s);                  // e = ea @ We
  fill_u32<<<(n_dst + 255) / 256, 256, 0, s>>>(S.mu, FENC_NEG_INF, n_dst);
  fill_f32<<<(n_dst + 255) / 256, 256, 0, s>>>(S.zb, 0.f, n_dst);
  edge_alpha<<<(E + 7) / 8, 256, 0, s>>>(S.qb, S.kb, S.eb, src, dst, E, S.al, S.mu);
  edge_exp<<<(E + 255) / 256, 256, 0, s>>>(S.al, dst, E, S.mu, S.wb, S.zb);
  gemm(xdst, Ws, bs, out, n_dst, Hc, accum_skip, s);            // skip into out
  edge_scatter<<<(E + 7) / 8, 256, 0, s>>>(S.vb, S.eb, src, dst, E, S.wb, S.zb, out);
}

extern "C" void kernel_launch(void* const* d_in, const int* in_sizes, int n_in,
                              void* d_out, int out_size, void* d_ws, size_t ws_size,
                              hipStream_t stream) {
  const float* user_emb = (const float*)d_in[0];
  const float* book_emb = (const float*)d_in[1];
  const float* eattr_ub = (const float*)d_in[2];
  const float* eattr_bu = (const float*)d_in[3];
  const float* eattr_bg = (const float*)d_in[4];
  const float* eattr_gb = (const float*)d_in[5];
  const float* Wq = (const float*)d_in[6];
  const float* bq = (const float*)d_in[7];
  const float* Wk = (const float*)d_in[8];
  const float* bk = (const float*)d_in[9];
  const float* Wv = (const float*)d_in[10];
  const float* bv = (const float*)d_in[11];
  const float* We = (const float*)d_in[12];
  const float* Wsk = (const float*)d_in[13];
  const float* bsk = (const float*)d_in[14];
  const int* user_nid  = (const int*)d_in[15];
  const int* book_nid  = (const int*)d_in[16];
  const int* genre_nid = (const int*)d_in[17];
  const int* src_ub = (const int*)d_in[18];
  const int* dst_ub = (const int*)d_in[19];
  const int* src_bu = (const int*)d_in[20];
  const int* dst_bu = (const int*)d_in[21];
  const int* src_bg = (const int*)d_in[22];
  const int* dst_bg = (const int*)d_in[23];
  const int* src_gb = (const int*)d_in[24];
  const int* dst_gb = (const int*)d_in[25];
  const int* label_src = (const int*)d_in[26];
  const int* label_dst = (const int*)d_in[27];
  (void)in_sizes; (void)n_in; (void)out_size; (void)ws_size;

  // carve workspace (floats); every row buffer is 128-float aligned -> float4 ok
  float* p = (float*)d_ws;
  auto take = [&](size_t n) { float* r = p; p += n; return r; };
  float* xu  = take((size_t)NUc * Hc);
  float* xb  = take((size_t)NBc * Hc);
  float* xg  = take((size_t)NGc * Hc);
  float* nxu = take((size_t)NUc * Hc);
  float* nxb = take((size_t)NBc * Hc);
  float* nxg = take((size_t)NGc * Hc);
  Scratch S;
  S.qb = take((size_t)NUc * Hc);
  S.kb = take((size_t)NUc * Hc);
  S.vb = take((size_t)NUc * Hc);
  S.eb = take((size_t)EUBc * Hc);
  S.al = take((size_t)EUBc);
  S.wb = take((size_t)EUBc);
  S.mu = (unsigned*)take((size_t)NUc);
  S.zb = take((size_t)NUc);

  // initial node features (faithful gathers, incl. genre <- book_emb)
  gather_rows<<<((NUc * Hc) + 255) / 256, 256, 0, stream>>>(user_emb, user_nid, xu, NUc);
  gather_rows<<<((NBc * Hc) + 255) / 256, 256, 0, stream>>>(book_emb, book_nid, xb, NBc);
  gather_rows<<<((NGc * Hc) + 255) / 256, 256, 0, stream>>>(book_emb, genre_nid, xg, NGc);

  float *cu = xu, *cb = xb, *cg = xg;
  float *ou = nxu, *ob = nxb, *og = nxg;

  for (int l = 0; l < Lc; ++l) {
    auto WQ = [&](int t) { return Wq  + (size_t)(l * 4 + t) * Hc * Hc; };
    auto BQ = [&](int t) { return bq  + (size_t)(l * 4 + t) * Hc; };
    auto WK = [&](int t) { return Wk  + (size_t)(l * 4 + t) * Hc * Hc; };
    auto BK = [&](int t) { return bk  + (size_t)(l * 4 + t) * Hc; };
    auto WV = [&](int t) { return Wv  + (size_t)(l * 4 + t) * Hc * Hc; };
    auto BV = [&](int t) { return bv  + (size_t)(l * 4 + t) * Hc; };
    auto WE = [&](int t) { return We  + (size_t)(l * 4 + t) * EDc * Hc; };
    auto WS = [&](int t) { return Wsk + (size_t)(l * 4 + t) * Hc * Hc; };
    auto BS = [&](int t) { return bsk + (size_t)(l * 4 + t) * Hc; };

    // nu = tconv(xb -> xu, bu edges, t=1)
    run_tconv(cb, cu, NBc, NUc, src_bu, dst_bu, EUBc, eattr_bu,
              WQ(1), BQ(1), WK(1), BK(1), WV(1), BV(1), WE(1), WS(1), BS(1),
              ou, 0, S, stream);
    // nb = tconv(xu -> xb, ub edges, t=0) ...
    run_tconv(cu, cb, NUc, NBc, src_ub, dst_ub, EUBc, eattr_ub,
              WQ(0), BQ(0), WK(0), BK(0), WV(0), BV(0), WE(0), WS(0), BS(0),
              ob, 0, S, stream);
    //      ... + tconv(xg -> xb, gb edges, t=3)  (HeteroConv sum)
    run_tconv(cg, cb, NGc, NBc, src_gb, dst_gb, EBGc, eattr_gb,
              WQ(3), BQ(3), WK(3), BK(3), WV(3), BV(3), WE(3), WS(3), BS(3),
              ob, 1, S, stream);
    // ng = tconv(xb -> xg, bg edges, t=2)
    run_tconv(cb, cg, NBc, NGc, src_bg, dst_bg, EBGc, eattr_bg,
              WQ(2), BQ(2), WK(2), BK(2), WV(2), BV(2), WE(2), WS(2), BS(2),
              og, 0, S, stream);

    if (l != Lc - 1) {
      relu_k<<<((NUc * Hc) + 255) / 256, 256, 0, stream>>>(ou, NUc * Hc);
      relu_k<<<((NBc * Hc) + 255) / 256, 256, 0, stream>>>(ob, NBc * Hc);
      relu_k<<<((NGc * Hc) + 255) / 256, 256, 0, stream>>>(og, NGc * Hc);
    }
    // swap current/next
    float* t;
    t = cu; cu = ou; ou = t;
    t = cb; cb = ob; ob = t;
    t = cg; cg = og; og = t;
  }

  label_dot<<<(NLc + 7) / 8, 256, 0, stream>>>(cu, cb, label_src, label_dst,
                                               (float*)d_out, NLc);
}
